// TypedAttributeEncoder_46901042872936
// MI455X (gfx1250) — matrix operand order repaired and checked by reading, not compiled, not run on hardware
//
#include <hip/hip_runtime.h>

typedef __attribute__((ext_vector_type(16))) __bf16 v16bf;
typedef __attribute__((ext_vector_type(8)))  float  v8f;

union BF16x16 {
  unsigned int u[8];
  v16bf v;
};

__device__ __forceinline__ unsigned short f2bf(float f) {
  unsigned int u = __float_as_uint(f);
  u += 0x7FFFu + ((u >> 16) & 1u);   // round-to-nearest-even bf16
  return (unsigned short)(u >> 16);
}
__device__ __forceinline__ float bf2f(unsigned short h) {
  return __uint_as_float(((unsigned int)h) << 16);
}
// Split two fp32 values into packed bf16 hi pair and bf16 lo (residual) pair.
__device__ __forceinline__ void split2(float a, float b,
                                       unsigned int& ph, unsigned int& pl) {
  unsigned short ha = f2bf(a), hb = f2bf(b);
  unsigned short la = f2bf(a - bf2f(ha)), lb = f2bf(b - bf2f(hb));
  ph = (unsigned int)ha | ((unsigned int)hb << 16);
  pl = (unsigned int)la | ((unsigned int)lb << 16);
}

__device__ __forceinline__ v8f wmma_bf16(v16bf a, v16bf b, v8f c) {
  // (neg_a, A, neg_b, B, c_mod, C, reuse_a, reuse_b)
  return __builtin_amdgcn_wmma_f32_16x16x32_bf16(false, a, false, b,
                                                 (short)0, c, false, false);
}

__device__ __forceinline__ void load_u8(const uint4* p0, const uint4* p1, BF16x16& f) {
  uint4 a = *p0, b = *p1;
  f.u[0] = a.x; f.u[1] = a.y; f.u[2] = a.z; f.u[3] = a.w;
  f.u[4] = b.x; f.u[5] = b.y; f.u[6] = b.z; f.u[7] = b.w;
}

struct TypeParams {
  const float* W1[4];           // [d,256] row-major fp32
  const float* b1[4];           // [256]
  const float* W2[4];           // [256,128] row-major fp32
  const float* b2[4];           // [128]
  const unsigned int* fW1[4];   // pre-split fragment-major (hi/lo) blocks
  const unsigned int* fW2[4];
};

#define HIDDEN 256
#define OUTD   128

// ---------------------------------------------------------------------------
// Phase 0: pre-split a fp32 weight matrix [K, ncols] into B-fragment-major
// bf16 hi/lo blocks. One block (= one 16x16 n-tile at one 32-deep k-tile) is
// 512 uints: [0..255] = hi packs (lane*8+j), [256..511] = lo packs.
// B layout (32x16): lane l -> col = l&15, K half = 16*(l>=16); VGPR j holds
// the pair (k0+2j, k0+2j+1).
// ---------------------------------------------------------------------------
__global__ __launch_bounds__(256)
void presplit_kernel(const float* __restrict__ W, unsigned int* __restrict__ dst,
                     int ncols) {
  const int f      = blockIdx.x;        // frag block id = kt*ntiles + nt
  const int ntiles = ncols >> 4;
  const int kt     = f / ntiles;
  const int nt     = f - kt * ntiles;
  const int tid    = threadIdx.x;       // == lane*8 + j
  const int lane   = tid >> 3;
  const int j      = tid & 7;
  const int col    = nt * 16 + (lane & 15);
  const int k      = kt * 32 + 16 * (lane >> 4) + 2 * j;
  unsigned int ph, pl;
  split2(W[k * ncols + col], W[(k + 1) * ncols + col], ph, pl);
  dst[f * 512 + tid]       = ph;
  dst[f * 512 + 256 + tid] = pl;
}

// Load one B fragment pair (hi, lo). PRE=true: 4x b128 from pre-split blocks.
// PRE=false: 16x b32 fp32 + inline split (fallback when ws is too small).
template <bool PRE>
__device__ __forceinline__ void load_bfrag(const float* __restrict__ W,
                                           const unsigned int* __restrict__ fW,
                                           int blk, int kk0, int ncols, int col,
                                           int lane, BF16x16& bh, BF16x16& bl) {
  if (PRE) {
    const uint4* p = reinterpret_cast<const uint4*>(fW + blk * 512 + lane * 8);
    const uint4* q = reinterpret_cast<const uint4*>(fW + blk * 512 + 256 + lane * 8);
    load_u8(p, p + 1, bh);
    load_u8(q, q + 1, bl);
  } else {
    #pragma unroll
    for (int j = 0; j < 8; ++j) {
      const float w0 = W[(kk0 + 2 * j)     * ncols + col];
      const float w1 = W[(kk0 + 2 * j + 1) * ncols + col];
      split2(w0, w1, bh.u[j], bl.u[j]);
    }
  }
}

template <bool PRE>
__global__ __launch_bounds__(256)
void typed_mlp_kernel(const float* __restrict__ x,
                      const int*   __restrict__ types,
                      TypeParams P,
                      float* __restrict__ out,
                      int n_nodes)
{
  __shared__ int s_cnt[4];
  __shared__ int s_list[4][256];
  // x tile in A-fragment-major layout: per k-tile a 512-uint block
  // ([0..255]=hi packs lane*8+j, [256..511]=lo packs). Max 4 k-tiles (d=128).
  __shared__ alignas(16) unsigned int s_xa[4 * 512];           // 8 KB
  __shared__ alignas(16) unsigned short s_hh[16 * HIDDEN];     // h as bf16 hi, 8 KB
  __shared__ alignas(16) unsigned short s_hl[16 * HIDDEN];     // h residual bf16, 8 KB

  const int tid  = threadIdx.x;
  const int base = blockIdx.x * 256;
  const int nb   = min(256, n_nodes - base);

  // ---- per-block type compaction (order-independent -> deterministic) ----
  if (tid < 4) s_cnt[tid] = 0;
  __syncthreads();
  if (tid < nb) {
    int t = types[base + tid];
    int slot = atomicAdd(&s_cnt[t], 1);
    s_list[t][slot] = base + tid;
  }
  __syncthreads();

  const int lane = tid & 31;
  const int wave = tid >> 5;
  const int ln   = lane & 15;   // column / row-in-half selector
  const int lh   = lane >> 4;   // which K/M half this lane covers

  const int dims[4] = {128, 64, 96, 128};

  for (int t = 0; t < 4; ++t) {
    const int cnt = s_cnt[t];
    const int d   = dims[t];
    const int nkt = d >> 5;
    const float* __restrict__ W1 = P.W1[t];
    const float* __restrict__ B1 = P.b1[t];
    const float* __restrict__ W2 = P.W2[t];
    const float* __restrict__ B2 = P.b2[t];
    const unsigned int* __restrict__ F1 = P.fW1[t];
    const unsigned int* __restrict__ F2 = P.fW2[t];
    const int ntiles = (cnt + 15) >> 4;

    for (int tile = 0; tile < ntiles; ++tile) {
      const int rbase = tile * 16;
      const int rows  = min(16, cnt - rbase);

      // ---- stage gathered x rows into LDS, A-fragment-major bf16 hi/lo ----
      // A layout (16x32): lane l -> row m = l&15; K base = 16*(l>=16);
      // VGPR j: pair at k_rel = 16*(l>=16) + 2j.
      for (int idx = tid; idx < nkt * 256; idx += 256) {
        const int kt    = idx >> 8;
        const int slot  = idx & 255;
        const int lane2 = slot >> 3;
        const int j     = slot & 7;
        const int m     = lane2 & 15;
        const int k     = kt * 32 + 16 * (lane2 >> 4) + 2 * j;
        unsigned int ph = 0u, pl = 0u;
        if (m < rows) {
          const int node = s_list[t][rbase + m];
          split2(x[node * 128 + k], x[node * 128 + k + 1], ph, pl);
        }
        s_xa[kt * 512 + slot]       = ph;
        s_xa[kt * 512 + 256 + slot] = pl;
      }
      __syncthreads();

      // ---- layer 1: h[16,256] = relu(x[16,d] @ W1 + b1) ----
      // wave w owns hidden columns [w*32, w*32+32) (two 16-wide n-tiles)
      v8f acc0 = {0,0,0,0,0,0,0,0};
      v8f acc1 = {0,0,0,0,0,0,0,0};
      for (int kb = 0; kb < d; kb += 32) {
        const int kt = kb >> 5;
        BF16x16 ah, al;
        {
          const uint4* ap = reinterpret_cast<const uint4*>(&s_xa[kt * 512 + lane * 8]);
          const uint4* lp = reinterpret_cast<const uint4*>(&s_xa[kt * 512 + 256 + lane * 8]);
          load_u8(ap, ap + 1, ah);
          load_u8(lp, lp + 1, al);
        }
        const int kk0 = kb + 16 * lh;
        #pragma unroll
        for (int nt = 0; nt < 2; ++nt) {
          const int col = wave * 32 + nt * 16 + ln;
          const int blk = kt * 16 + wave * 2 + nt;
          BF16x16 bh, bl;
          load_bfrag<PRE>(W1, F1, blk, kk0, HIDDEN, col, lane, bh, bl);
          v8f c = nt ? acc1 : acc0;
          c = wmma_bf16(ah.v, bh.v, c);   // hi*hi
          c = wmma_bf16(ah.v, bl.v, c);   // hi*lo
          c = wmma_bf16(al.v, bh.v, c);   // lo*hi
          if (nt) acc1 = c; else acc0 = c;
        }
      }
      // bias + relu; store h as bf16 hi/lo shorts (C layout: VGPR v -> M=v+8*lh, N=col)
      #pragma unroll
      for (int nt = 0; nt < 2; ++nt) {
        const int col  = wave * 32 + nt * 16 + ln;
        const float bs = B1[col];
        v8f c = nt ? acc1 : acc0;
        #pragma unroll
        for (int v = 0; v < 8; ++v) {
          const int m = v + 8 * lh;
          const float val = fmaxf(c[v] + bs, 0.0f);
          const unsigned short hh = f2bf(val);
          const unsigned short hl = f2bf(val - bf2f(hh));
          s_hh[m * HIDDEN + col] = hh;
          s_hl[m * HIDDEN + col] = hl;
        }
      }
      __syncthreads();

      // ---- layer 2: out[16,128] = h[16,256] @ W2 + b2; wave w -> cols [w*16, w*16+16) ----
      // A fragments read straight from the pair-packed short arrays:
      // VGPRs 0-3 = 16 contiguous bytes at (ln,k0), VGPRs 4-7 at (ln,k0+8), k0 = kb+16*lh.
      v8f oc = {0,0,0,0,0,0,0,0};
      const int col2 = wave * 16 + ln;
      #pragma unroll
      for (int kb = 0; kb < HIDDEN; kb += 32) {
        const int k0 = kb + 16 * lh;
        BF16x16 ah, al;
        {
          const uint4* hp = reinterpret_cast<const uint4*>(&s_hh[ln * HIDDEN + k0]);
          const uint4* lp = reinterpret_cast<const uint4*>(&s_hl[ln * HIDDEN + k0]);
          load_u8(hp, hp + 1, ah);
          load_u8(lp, lp + 1, al);
        }
        const int blk = (kb >> 5) * 8 + wave;
        BF16x16 bh, bl;
        load_bfrag<PRE>(W2, F2, blk, kb + 16 * lh, OUTD, col2, lane, bh, bl);
        oc = wmma_bf16(ah.v, bh.v, oc);
        oc = wmma_bf16(ah.v, bl.v, oc);
        oc = wmma_bf16(al.v, bh.v, oc);
      }
      const float bs2 = B2[col2];
      // scatter result rows back by node id (skip padding rows)
      #pragma unroll
      for (int v = 0; v < 8; ++v) {
        const int m = v + 8 * lh;
        if (m < rows) {
          const int node = s_list[t][rbase + m];
          out[node * OUTD + col2] = oc[v] + bs2;
        }
      }
      __syncthreads();
    }
  }
}

extern "C" void kernel_launch(void* const* d_in, const int* in_sizes, int n_in,
                              void* d_out, int out_size, void* d_ws, size_t ws_size,
                              hipStream_t stream) {
  (void)n_in; (void)out_size;
  const float* x   = (const float*)d_in[0];
  const int* types = (const int*)d_in[1];
  static const int dims[4] = {128, 64, 96, 128};

  TypeParams P;
  for (int t = 0; t < 4; ++t) {
    P.W1[t] = (const float*)d_in[2 + 4 * t];
    P.b1[t] = (const float*)d_in[3 + 4 * t];
    P.W2[t] = (const float*)d_in[4 + 4 * t];
    P.b2[t] = (const float*)d_in[5 + 4 * t];
    P.fW1[t] = nullptr;
    P.fW2[t] = nullptr;
  }

  // ws layout: per type, W1 frags ((d/32)*16 blocks) then W2 frags (64 blocks),
  // each block 512 uints. Total = 237568 uints = 950272 bytes.
  size_t need_uints = 0;
  for (int t = 0; t < 4; ++t) need_uints += (size_t)(dims[t] / 32) * 16 * 512 + 64 * 512;
  const bool pre = (d_ws != nullptr) && (ws_size >= need_uints * 4);

  const int n = in_sizes[1];               // N_NODES (node_types count)
  const int blocks = (n + 255) / 256;

  if (pre) {
    unsigned int* ws = (unsigned int*)d_ws;
    size_t off = 0;
    unsigned int* f1[4]; unsigned int* f2[4];
    for (int t = 0; t < 4; ++t) {
      f1[t] = ws + off; off += (size_t)(dims[t] / 32) * 16 * 512;
      f2[t] = ws + off; off += 64 * 512;
      P.fW1[t] = f1[t];
      P.fW2[t] = f2[t];
    }
    for (int t = 0; t < 4; ++t) {
      presplit_kernel<<<(dims[t] / 32) * 16, 256, 0, stream>>>(P.W1[t], f1[t], HIDDEN);
      presplit_kernel<<<8 * 8,              256, 0, stream>>>(P.W2[t], f2[t], OUTD);
    }
    typed_mlp_kernel<true><<<blocks, 256, 0, stream>>>(x, types, P, (float*)d_out, n);
  } else {
    typed_mlp_kernel<false><<<blocks, 256, 0, stream>>>(x, types, P, (float*)d_out, n);
  }
}
